// AffinityPosweight_61830349193486
// MI455X (gfx1250) — matrix operand order repaired and checked by reading, not compiled, not run on hardware
//
#include <hip/hip_runtime.h>
#include <hip/hip_bf16.h>

typedef __attribute__((ext_vector_type(16))) __bf16 v16bf;
typedef __attribute__((ext_vector_type(8)))  float  v8f;

#define SEQ_L 4096
#define DIM_D 1024
#define DIM_H 1024

// ---------------------------------------------------------------------------
// WMMA fragment loaders (wave32, v_wmma_f32_16x16x32_bf16 layouts per ISA):
//  A (16x32, MxK): lane = half*16 + m ; e[i] = A[m][ (i/8)*16 + half*8 + i%8 ]
//    -> two contiguous 16-byte loads per lane.
//  B (32x16, KxN): lane = half*16 + n ; e[i] = B[ (i/8)*16 + half*8 + i%8 ][ n ]
//    -> with the operand stored N-major (Bt[n][k]), two contiguous 16-byte loads.
// ---------------------------------------------------------------------------
__device__ __forceinline__ v16bf load_a_frag(const __bf16* A, int lda,
                                             int row0, int k0, int m, int half) {
  union { v16bf v; __bf16 e[16]; } u;
  const __bf16* p = A + (size_t)(row0 + m) * lda + k0 + half * 8;
  __builtin_memcpy(&u.e[0], p,      16);   // K = k0 + half*8 + 0..7
  __builtin_memcpy(&u.e[8], p + 16, 16);   // K = k0 + 16 + half*8 + 0..7
  return u.v;
}

__device__ __forceinline__ v16bf load_b_frag(const __bf16* Bt, int ldb,
                                             int n0, int k0, int n, int half) {
  union { v16bf v; __bf16 e[16]; } u;
  const __bf16* p = Bt + (size_t)(n0 + n) * ldb + k0 + half * 8;
  __builtin_memcpy(&u.e[0], p,      16);   // K = k0 + half*8 + 0..7
  __builtin_memcpy(&u.e[8], p + 16, 16);   // K = k0 + 16 + half*8 + 0..7
  return u.v;
}

#define WMMA_BF16(a, b, c) \
  __builtin_amdgcn_wmma_f32_16x16x32_bf16(false, (a), false, (b), (short)0, (c), false, false)

// ---------------------------------------------------------------------------
// Prep: fp32 -> bf16 convert, and transposing convert for weight matrices.
// ---------------------------------------------------------------------------
__global__ __launch_bounds__(256) void cvt_f32_bf16(const float* __restrict__ src,
                                                    __bf16* __restrict__ dst, int n) {
  int i = blockIdx.x * 256 + threadIdx.x;
  if (i < n) dst[i] = (__bf16)src[i];
}

// W: [D][H] fp32 row-major  ->  Wt: [H][D] bf16 row-major
__global__ __launch_bounds__(256) void transpose_cvt(const float* __restrict__ W,
                                                     __bf16* __restrict__ Wt) {
  int idx = blockIdx.x * 256 + threadIdx.x;   // over D*H
  int k = idx / DIM_H, n = idx % DIM_H;
  Wt[(size_t)n * DIM_D + k] = (__bf16)W[(size_t)k * DIM_H + n];
}

// ---------------------------------------------------------------------------
// Core register-blocked GEMM body: one wave computes a 32x64 output tile.
// Per k-step: 2 A-frags + 4 B-frags feed 8 WMMAs (0.75 fragment loads/WMMA).
// Accumulators: 8 x v8f = 64 VGPRs.
// ---------------------------------------------------------------------------
struct Acc32x64 { v8f c[2][4]; };

__device__ __forceinline__ void gemm_step(Acc32x64& acc,
                                          const __bf16* A, int lda, int row0,
                                          const __bf16* Bt, int ldb, int n0,
                                          int k0, int m, int half) {
  v16bf a0 = load_a_frag(A, lda, row0,      k0, m, half);
  v16bf a1 = load_a_frag(A, lda, row0 + 16, k0, m, half);
#pragma unroll
  for (int j = 0; j < 4; ++j) {
    v16bf b = load_b_frag(Bt, ldb, n0 + 16 * j, k0, m, half);
    acc.c[0][j] = WMMA_BF16(a0, b, acc.c[0][j]);
    acc.c[1][j] = WMMA_BF16(a1, b, acc.c[1][j]);
  }
}

// ---------------------------------------------------------------------------
// Kernel 1: Q = x Wq, K = x Wk, V = x Wv  (V stored transposed, H x L)
// one 32x64 output tile per wave, K-loop = 1024 -> 256 WMMAs per wave
// ---------------------------------------------------------------------------
__global__ __launch_bounds__(256) void qkv_gemm(const __bf16* __restrict__ xb,
                                                const __bf16* __restrict__ wqt,
                                                const __bf16* __restrict__ wkt,
                                                const __bf16* __restrict__ wvt,
                                                __bf16* __restrict__ qb,
                                                __bf16* __restrict__ kb,
                                                __bf16* __restrict__ vt) {
  const int wid  = blockIdx.x * 8 + (threadIdx.x >> 5);
  const int lane = threadIdx.x & 31;
  const int m    = lane & 15;
  const int half = lane >> 4;

  const int mat  = wid / (128 * 16);         // 0=Q 1=K 2=V
  const int rem  = wid % (128 * 16);
  const int row0 = (rem / 16) * 32;          // L/32 = 128 row tiles
  const int n0   = (rem % 16) * 64;          // H/64 = 16 col tiles
  const __bf16* Bt = (mat == 0) ? wqt : (mat == 1) ? wkt : wvt;

  Acc32x64 acc = {};
#pragma unroll 2
  for (int k0 = 0; k0 < DIM_D; k0 += 32) {
    __builtin_prefetch(xb + (size_t)(row0 + m) * DIM_D + k0 + 64, 0, 1);
    __builtin_prefetch(Bt + (size_t)(n0 + m) * DIM_D + k0 + 64, 0, 1);
    gemm_step(acc, xb, DIM_D, row0, Bt, DIM_D, n0, k0, m, half);
  }

  if (mat == 2) {
#pragma unroll
    for (int i2 = 0; i2 < 2; ++i2)
#pragma unroll
      for (int j = 0; j < 4; ++j)
#pragma unroll
        for (int r = 0; r < 8; ++r)          // V^T: vt[h][l]
          vt[(size_t)(n0 + 16 * j + m) * SEQ_L + row0 + 16 * i2 + half * 8 + r] =
              (__bf16)acc.c[i2][j][r];
  } else {
    __bf16* Obuf = (mat == 0) ? qb : kb;
#pragma unroll
    for (int i2 = 0; i2 < 2; ++i2)
#pragma unroll
      for (int j = 0; j < 4; ++j)
#pragma unroll
        for (int r = 0; r < 8; ++r)
          Obuf[(size_t)(row0 + 16 * i2 + half * 8 + r) * DIM_H + n0 + 16 * j + m] =
              (__bf16)acc.c[i2][j][r];
  }
}

// ---------------------------------------------------------------------------
// Kernel 2: S = (Q K^T) / sqrt(H), tiles kept iff any column <= max row
// (2*tj <= ti). K^T is the natural B operand of row-major K.
// ---------------------------------------------------------------------------
__global__ __launch_bounds__(256) void score_gemm(const __bf16* __restrict__ qb,
                                                  const __bf16* __restrict__ kb,
                                                  __bf16* __restrict__ sp) {
  const int wid  = blockIdx.x * 8 + (threadIdx.x >> 5);
  const int lane = threadIdx.x & 31;
  const int m    = lane & 15;
  const int half = lane >> 4;
  const int ti   = wid >> 6;                 // query tile (32 rows), 0..127
  const int tj   = wid & 63;                 // key tile (64 cols),  0..63
  if (2 * tj > ti) return;                   // wave-uniform causal skip
  const int row0 = ti * 32, n0 = tj * 64;

  Acc32x64 acc = {};
#pragma unroll 2
  for (int k0 = 0; k0 < DIM_H; k0 += 32) {
    __builtin_prefetch(qb + (size_t)(row0 + m) * DIM_H + k0 + 64, 0, 1);
    __builtin_prefetch(kb + (size_t)(n0 + m) * DIM_H + k0 + 64, 0, 1);
    gemm_step(acc, qb, DIM_H, row0, kb, DIM_H, n0, k0, m, half);
  }
#pragma unroll
  for (int i2 = 0; i2 < 2; ++i2)
#pragma unroll
    for (int j = 0; j < 4; ++j)
#pragma unroll
      for (int r = 0; r < 8; ++r)            // 1/sqrt(1024) = 1/32
        sp[(size_t)(row0 + 16 * i2 + half * 8 + r) * SEQ_L + n0 + 16 * j + m] =
            (__bf16)(acc.c[i2][j][r] * 0.03125f);
}

// ---------------------------------------------------------------------------
// Kernel 3: row softmax (fp32 math), in-place bf16. One block per row.
// Zero-pads [i+1, roundup32(i+1)) so the PV kernel's 32-aligned k-loop is safe
// — this is where the causal mask is realized (no mask matrix read at all).
// ---------------------------------------------------------------------------
__global__ __launch_bounds__(256) void softmax_rows(__bf16* __restrict__ sp) {
  const int i      = blockIdx.x;
  __bf16*   row    = sp + (size_t)i * SEQ_L;
  const int nvalid = i + 1;
  const int limit  = ((i >> 5) + 1) << 5;    // roundup32(i+1)
  const int t      = threadIdx.x;
  __shared__ float red[256];

  float mx = -3.0e38f;
  for (int j = t; j < nvalid; j += 256) mx = fmaxf(mx, (float)row[j]);
  red[t] = mx; __syncthreads();
  for (int s = 128; s > 0; s >>= 1) {
    if (t < s) red[t] = fmaxf(red[t], red[t + s]);
    __syncthreads();
  }
  mx = red[0]; __syncthreads();

  float sum = 0.f;
  for (int j = t; j < nvalid; j += 256) sum += __expf((float)row[j] - mx);
  red[t] = sum; __syncthreads();
  for (int s = 128; s > 0; s >>= 1) {
    if (t < s) red[t] += red[t + s];
    __syncthreads();
  }
  const float inv = 1.0f / red[0];

  for (int j = t; j < limit; j += 256) {
    float p = (j < nvalid) ? __expf((float)row[j] - mx) * inv : 0.f;
    row[j] = (__bf16)p;
  }
}

// ---------------------------------------------------------------------------
// Kernel 4: out = P V. k-loop bounded by causal extent of the 32-row tile:
// kmax = (ti+1)*32, which equals softmax's zero-pad limit for every row in
// the tile (halves the flops, never reads unwritten memory).
// ---------------------------------------------------------------------------
__global__ __launch_bounds__(256) void out_gemm(const __bf16* __restrict__ sp,
                                                const __bf16* __restrict__ vt,
                                                float* __restrict__ out) {
  const int wid  = blockIdx.x * 8 + (threadIdx.x >> 5);
  const int lane = threadIdx.x & 31;
  const int m    = lane & 15;
  const int half = lane >> 4;
  const int ti   = wid >> 4;                 // row tile (32 rows), 0..127
  const int tj   = wid & 15;                 // col tile (64 cols), 0..15
  const int row0 = ti * 32, n0 = tj * 64;
  const int kmax = (ti + 1) * 32;

  Acc32x64 acc = {};
  for (int k0 = 0; k0 < kmax; k0 += 32) {
    gemm_step(acc, sp, SEQ_L, row0, vt, SEQ_L, n0, k0, m, half);
  }
#pragma unroll
  for (int i2 = 0; i2 < 2; ++i2)
#pragma unroll
    for (int j = 0; j < 4; ++j)
#pragma unroll
      for (int r = 0; r < 8; ++r)
        out[(size_t)(row0 + 16 * i2 + half * 8 + r) * DIM_H + n0 + 16 * j + m] =
            acc.c[i2][j][r];
}

// ---------------------------------------------------------------------------
// Host launcher. ws layout (bytes):
//   xb 0..8M | Wqt 8..10M | Wkt 10..12M | Wvt 12..14M |
//   qb 14..22M | kb 22..30M | vt 30..38M | sp 38..70M
// ---------------------------------------------------------------------------
extern "C" void kernel_launch(void* const* d_in, const int* in_sizes, int n_in,
                              void* d_out, int out_size, void* d_ws, size_t ws_size,
                              hipStream_t stream) {
  (void)in_sizes; (void)n_in; (void)out_size; (void)ws_size;
  const float* x  = (const float*)d_in[0];
  // d_in[1] = mask: unused — causality handled analytically by loop bounds.
  const float* Wq = (const float*)d_in[2];
  const float* Wk = (const float*)d_in[3];
  const float* Wv = (const float*)d_in[4];

  char* ws = (char*)d_ws;
  const size_t MB = 1024u * 1024u;
  __bf16* xb  = (__bf16*)(ws);
  __bf16* wqt = (__bf16*)(ws +  8 * MB);
  __bf16* wkt = (__bf16*)(ws + 10 * MB);
  __bf16* wvt = (__bf16*)(ws + 12 * MB);
  __bf16* qb  = (__bf16*)(ws + 14 * MB);
  __bf16* kb  = (__bf16*)(ws + 22 * MB);
  __bf16* vt  = (__bf16*)(ws + 30 * MB);
  __bf16* sp  = (__bf16*)(ws + 38 * MB);

  cvt_f32_bf16<<<(SEQ_L * DIM_D) / 256, 256, 0, stream>>>(x, xb, SEQ_L * DIM_D);
  transpose_cvt<<<(DIM_D * DIM_H) / 256, 256, 0, stream>>>(Wq, wqt);
  transpose_cvt<<<(DIM_D * DIM_H) / 256, 256, 0, stream>>>(Wk, wkt);
  transpose_cvt<<<(DIM_D * DIM_H) / 256, 256, 0, stream>>>(Wv, wvt);

  qkv_gemm<<<(3 * 128 * 16) / 8, 256, 0, stream>>>(xb, wqt, wkt, wvt, qb, kb, vt);
  score_gemm<<<(128 * 64) / 8, 256, 0, stream>>>(qb, kb, sp);
  softmax_rows<<<SEQ_L, 256, 0, stream>>>(sp);
  out_gemm<<<(128 * 16) / 8, 256, 0, stream>>>(sp, vt, (float*)d_out);
}
